// Baseline_25383256719506
// MI455X (gfx1250) — compile-verified, matrix operand order checked by reading, not run on hardware
//
#include <hip/hip_runtime.h>

typedef __attribute__((ext_vector_type(4)))  float  v4f;
typedef __attribute__((ext_vector_type(8)))  float  v8f;
typedef __attribute__((ext_vector_type(4)))  __bf16 v4bf;
typedef __attribute__((ext_vector_type(8)))  __bf16 v8bf;
typedef __attribute__((ext_vector_type(16))) __bf16 v16bf;

// ---------------------------------------------------------------------------
// f32 -> bf16 convert (4 elems / thread): input x, agg buffers.
// ---------------------------------------------------------------------------
__global__ __launch_bounds__(256)
void conv_bf16_kernel(const float* __restrict__ in, __bf16* __restrict__ out,
                      long long n4) {
    long long i = (long long)blockIdx.x * blockDim.x + threadIdx.x;
    if (i >= n4) return;
    v4f v = *(const v4f*)(in + i * 4);
    v4bf o;
    o[0] = (__bf16)v.x; o[1] = (__bf16)v.y; o[2] = (__bf16)v.z; o[3] = (__bf16)v.w;
    *(v4bf*)(out + i * 4) = o;
}

// Weight prep: W[k][n] (f32) -> Wt[n][k] (bf16): B-operand loads contiguous in K.
__global__ void wt_bf16_kernel(const float* __restrict__ W, __bf16* __restrict__ Wt,
                               int K, int Nc) {
    int tid = blockIdx.x * blockDim.x + threadIdx.x;
    if (tid >= K * Nc) return;
    int k = tid / Nc, n = tid % Nc;
    Wt[(size_t)n * K + k] = (__bf16)W[(size_t)k * Nc + n];
}

__global__ void zero_kernel(float* __restrict__ p, long long n) {
    long long i = (long long)blockIdx.x * blockDim.x + threadIdx.x;
    if (i < n) p[i] = 0.0f;
}

// ---------------------------------------------------------------------------
// Edge scatter-add: agg[dst] += feat[src]. bf16 gather (half the read traffic),
// f32 global atomics for the accumulation. One thread per (edge, 4 features).
// ---------------------------------------------------------------------------
__global__ __launch_bounds__(256)
void scatter_add_kernel(const __bf16* __restrict__ feat, float* __restrict__ agg,
                        const long long* __restrict__ ei, int E, int F, int fshift) {
    long long tid = (long long)blockIdx.x * blockDim.x + threadIdx.x;
    int e = (int)(tid >> fshift);
    if (e >= E) return;
    int c = (int)((tid & ((1 << fshift) - 1)) << 2);
    int s = (int)ei[e];
    int d = (int)ei[E + e];
    v4bf v = *(const v4bf*)(feat + (size_t)s * F + c);
    float* dp = agg + (size_t)d * F + c;
    atomicAdd(dp + 0, (float)v[0]);
    atomicAdd(dp + 1, (float)v[1]);
    atomicAdd(dp + 2, (float)v[2]);
    atomicAdd(dp + 3, (float)v[3]);
}

// ---------------------------------------------------------------------------
// WMMA: one wave owns a 16x64 supertile (4 accumulators). A fragment is loaded
// once per K-step and reused across 4 WMMAs (2.5 b128 loads per WMMA).
// Fragment layout per ISA 16-bit 16x32 map: lane&15 = row/col, half = lane>>4
// selects K sub-bands {h*8..h*8+7, 16+h*8..16+h*8+7}.
// ---------------------------------------------------------------------------
static __device__ __forceinline__
void wmma_quad(v8f* acc, const __bf16* __restrict__ a_row,
               const __bf16* __restrict__ b_base, size_t bstride, int K, int half) {
    for (int k0 = 0; k0 < K; k0 += 32) {
        const int kb = k0 + half * 8;
        v8bf a0 = *(const v8bf*)(a_row + kb);
        v8bf a1 = *(const v8bf*)(a_row + kb + 16);
        v16bf a = __builtin_shufflevector(a0, a1, 0, 1, 2, 3, 4, 5, 6, 7,
                                          8, 9, 10, 11, 12, 13, 14, 15);
#pragma unroll
        for (int q = 0; q < 4; ++q) {
            const __bf16* bc = b_base + (size_t)q * bstride;
            v8bf b0 = *(const v8bf*)(bc + kb);
            v8bf b1 = *(const v8bf*)(bc + kb + 16);
            v16bf b = __builtin_shufflevector(b0, b1, 0, 1, 2, 3, 4, 5, 6, 7,
                                              8, 9, 10, 11, 12, 13, 14, 15);
            acc[q] = __builtin_amdgcn_wmma_f32_16x16x32_bf16(
                /*neg_a=*/false, a, /*neg_b=*/false, b,
                /*c_mod=*/(short)0, acc[q], /*reuse_a=*/false, /*reuse_b=*/false);
        }
    }
}

// ---------------------------------------------------------------------------
// Fused GraphConv GEMM: out_bf16 = relu?( A1@W1 [+ A2@W2] [+ P_f32] [+ bias] )
// 16x64 supertile per wave, 8 waves / 256-thread block (wave32). All dout
// values (64/128/192) are multiples of 64; nrows (50000) is a multiple of 16.
// ---------------------------------------------------------------------------
__global__ __launch_bounds__(256)
void gemm_bf16_wmma(const __bf16* __restrict__ A1, const __bf16* __restrict__ W1, int K1,
                    const __bf16* __restrict__ A2, const __bf16* __restrict__ W2, int K2,
                    const float* __restrict__ P, const float* __restrict__ bias,
                    __bf16* __restrict__ out, int nrows, int dout, int do_relu) {
    const int col_super = dout >> 6;
    const int row_tiles = nrows >> 4;
    int wg = blockIdx.x * (blockDim.x >> 5) + (threadIdx.x >> 5);
    if (wg >= row_tiles * col_super) return;
    const int rt = wg / col_super;      // consecutive waves share the row strip
    const int cs = wg % col_super;
    const int lane = threadIdx.x & 31;
    const int half = lane >> 4;
    const int l15  = lane & 15;
    const int row  = rt * 16 + l15;
    const int colb = cs * 64 + l15;     // B strip base column for this lane

    v8f acc[4];
#pragma unroll
    for (int q = 0; q < 4; ++q) acc[q] = (v8f){0.f, 0.f, 0.f, 0.f, 0.f, 0.f, 0.f, 0.f};

    wmma_quad(acc, A1 + (size_t)row * K1, W1 + (size_t)colb * K1, (size_t)16 * K1, K1, half);
    if (A2)
        wmma_quad(acc, A2 + (size_t)row * K2, W2 + (size_t)colb * K2, (size_t)16 * K2, K2, half);

#pragma unroll
    for (int q = 0; q < 4; ++q) {
        const int col = cs * 64 + q * 16 + l15;
        const float bval = bias ? bias[col] : 0.0f;   // whole lane shares one column
#pragma unroll
        for (int v = 0; v < 8; ++v) {
            const int r = rt * 16 + 8 * half + v;     // ISA C/D layout: M = v + 8*half
            float val = acc[q][v] + bval;
            if (P) val += P[(size_t)r * dout + col];
            if (do_relu) val = fmaxf(val, 0.0f);
            out[(size_t)r * dout + col] = (__bf16)val;
        }
    }
}

// ---------------------------------------------------------------------------
// Pooling + head (H = 64 fixed). Sums accumulated in f32.
// ---------------------------------------------------------------------------
__global__ __launch_bounds__(256)
void pool_sum_kernel(const __bf16* __restrict__ h, const long long* __restrict__ batch,
                     float* __restrict__ sums, int Nn) {
    int tid = blockIdx.x * blockDim.x + threadIdx.x;
    if (tid >= Nn * 16) return;                   // 16 x 4-feature chunks per node
    int n = tid >> 4;
    int c = (tid & 15) << 2;
    int g = (int)batch[n];
    v4bf v = *(const v4bf*)(h + (size_t)n * 64 + c);
    float* dp = sums + (size_t)g * 64 + c;
    atomicAdd(dp + 0, (float)v[0]);
    atomicAdd(dp + 1, (float)v[1]);
    atomicAdd(dp + 2, (float)v[2]);
    atomicAdd(dp + 3, (float)v[3]);
}

__global__ void count_kernel(const long long* __restrict__ batch, float* __restrict__ counts,
                             int Nn) {
    int n = blockIdx.x * blockDim.x + threadIdx.x;
    if (n >= Nn) return;
    atomicAdd(&counts[(int)batch[n]], 1.0f);
}

__global__ void head_kernel(const float* __restrict__ sums, const float* __restrict__ counts,
                            const float* __restrict__ hw, const float* __restrict__ hb,
                            float* __restrict__ out, int G) {
    int g = blockIdx.x * blockDim.x + threadIdx.x;
    if (g >= G) return;
    float c = fmaxf(counts[g], 1.0f);
    float acc = 0.0f;
    for (int i = 0; i < 64; ++i) acc += sums[(size_t)g * 64 + i] * hw[i];
    out[g] = acc / c + hb[0];
}

// ---------------------------------------------------------------------------
// Host orchestration
// ---------------------------------------------------------------------------
extern "C" void kernel_launch(void* const* d_in, const int* in_sizes, int n_in,
                              void* d_out, int out_size, void* d_ws, size_t ws_size,
                              hipStream_t stream) {
    const float*     x      = (const float*)d_in[0];
    const long long* ei     = (const long long*)d_in[1];
    const long long* batch  = (const long long*)d_in[2];
    const float* w1_rel  = (const float*)d_in[3];
    const float* w1_root = (const float*)d_in[4];
    const float* b1      = (const float*)d_in[5];
    const float* w2_rel  = (const float*)d_in[6];
    const float* w2_root = (const float*)d_in[7];
    const float* b2      = (const float*)d_in[8];
    const float* w3_rel  = (const float*)d_in[9];
    const float* w3_root = (const float*)d_in[10];
    const float* b3      = (const float*)d_in[11];
    const float* w4_rel  = (const float*)d_in[12];
    const float* w4_root = (const float*)d_in[13];
    const float* b4      = (const float*)d_in[14];
    const float* head_w  = (const float*)d_in[15];
    const float* head_b  = (const float*)d_in[16];

    const int Nn = in_sizes[0] / 128;   // 50000 (divisible by 16)
    const int E  = in_sizes[1] / 2;     // 800000
    const int G  = out_size;            // 64
    float* out = (float*)d_out;

    char* ws = (char*)d_ws;
    size_t off = 0;
    auto alloc = [&](size_t bytes) -> void* {
        void* p = ws + off;
        off += (bytes + 255) & ~(size_t)255;
        return p;
    };
    __bf16* Xb  = (__bf16*)alloc((size_t)Nn * 128 * 2);  // bf16 copy of x
    __bf16* T1b = (__bf16*)alloc((size_t)Nn * 64  * 2);  // y1 / h1 / y4
    float*  T2  = (float*)alloc((size_t)Nn * 128 * 4);   // f32 agg (atomic target)
    __bf16* T2b = (__bf16*)alloc((size_t)Nn * 128 * 2);  // bf16 copy of agg
    __bf16* T3b = (__bf16*)alloc((size_t)Nn * 128 * 2);  // h2 / h4
    __bf16* T4b = (__bf16*)alloc((size_t)Nn * 192 * 2);  // h3
    float* sums   = (float*)alloc((size_t)G * 64 * 4);
    float* counts = (float*)alloc((size_t)G * 4);
    __bf16* w1r = (__bf16*)alloc(128 * 64  * 2);
    __bf16* w1q = (__bf16*)alloc(128 * 64  * 2);
    __bf16* w2r = (__bf16*)alloc(64  * 128 * 2);
    __bf16* w2q = (__bf16*)alloc(64  * 128 * 2);
    __bf16* w3r = (__bf16*)alloc(128 * 192 * 2);
    __bf16* w3q = (__bf16*)alloc(128 * 192 * 2);
    __bf16* w4r = (__bf16*)alloc(192 * 64  * 2);
    __bf16* w4q = (__bf16*)alloc(192 * 64  * 2);
    (void)ws_size; (void)n_in;

    auto zero = [&](float* p, long long n) {
        zero_kernel<<<(unsigned)((n + 255) / 256), 256, 0, stream>>>(p, n);
    };
    auto conv = [&](const float* in, __bf16* o, long long n) {  // n multiple of 4
        long long n4 = n >> 2;
        conv_bf16_kernel<<<(unsigned)((n4 + 255) / 256), 256, 0, stream>>>(in, o, n4);
    };
    auto wt = [&](const float* W, __bf16* Wt, int K, int Nc) {
        int t = K * Nc;
        wt_bf16_kernel<<<(t + 255) / 256, 256, 0, stream>>>(W, Wt, K, Nc);
    };
    auto gemm = [&](const __bf16* A1, const __bf16* W1, int K1,
                    const __bf16* A2, const __bf16* W2, int K2,
                    const float* P, const float* bias, __bf16* o, int dout, int relu) {
        int waves  = (Nn / 16) * (dout / 64);
        int blocks = (waves + 7) / 8;
        gemm_bf16_wmma<<<blocks, 256, 0, stream>>>(A1, W1, K1, A2, W2, K2, P, bias,
                                                   o, Nn, dout, relu);
    };
    auto scatter = [&](const __bf16* f, float* agg, int F) {
        int fshift = (F == 64) ? 4 : 5;
        long long total = (long long)E << fshift;
        scatter_add_kernel<<<(unsigned)((total + 255) / 256), 256, 0, stream>>>(
            f, agg, ei, E, F, fshift);
    };

    // Prep: bf16 input copy + bf16 transposed weights
    conv(x, Xb, (long long)Nn * 128);
    wt(w1_rel, w1r, 128, 64);  wt(w1_root, w1q, 128, 64);
    wt(w2_rel, w2r, 64, 128);  wt(w2_root, w2q, 64, 128);
    wt(w3_rel, w3r, 128, 192); wt(w3_root, w3q, 128, 192);
    wt(w4_rel, w4r, 192, 64);  wt(w4_root, w4q, 192, 64);

    // Layer 1 (128 -> 64): reorder (GEMM before scatter: E*64 traffic, not E*128)
    gemm(Xb, w1r, 128, nullptr, nullptr, 0, nullptr, nullptr, T1b, 64, 0);  // y1
    zero(T2, (long long)Nn * 64);
    scatter(T1b, T2, 64);                                                   // agg1
    gemm(Xb, w1q, 128, nullptr, nullptr, 0, T2, b1, T1b, 64, 1);            // h1 = relu(agg1 + x@W1root + b1)

    // Layer 2 (64 -> 128): original order (din=64 < dout=128)
    zero(T2, (long long)Nn * 64);
    scatter(T1b, T2, 64);                                                   // agg2
    conv(T2, T2b, (long long)Nn * 64);
    gemm(T2b, w2r, 64, T1b, w2q, 64, nullptr, b2, T3b, 128, 1);             // h2

    // Layer 3 (128 -> 192): original order
    zero(T2, (long long)Nn * 128);
    scatter(T3b, T2, 128);                                                  // agg3
    conv(T2, T2b, (long long)Nn * 128);
    gemm(T2b, w3r, 128, T3b, w3q, 128, nullptr, b3, T4b, 192, 1);           // h3

    // Layer 4 (192 -> 64): reorder (E*64 traffic, not E*192)
    gemm(T4b, w4r, 192, nullptr, nullptr, 0, nullptr, nullptr, T1b, 64, 0); // y4
    zero(T2, (long long)Nn * 64);
    scatter(T1b, T2, 64);                                                   // agg4
    gemm(T4b, w4q, 192, nullptr, nullptr, 0, T2, b4, T3b, 64, 0);           // h4 (no relu)

    // Mean pool + head
    zero(sums, (long long)G * 64);
    zero(counts, G);
    pool_sum_kernel<<<(Nn * 16 + 255) / 256, 256, 0, stream>>>(T3b, batch, sums, Nn);
    count_kernel<<<(Nn + 255) / 256, 256, 0, stream>>>(batch, counts, Nn);
    head_kernel<<<1, 64, 0, stream>>>(sums, counts, head_w, head_b, out, G);
}
